// PRDCLoss_82652350644511
// MI455X (gfx1250) — matrix operand order repaired and checked by reading, not compiled, not run on hardware
//
#include <hip/hip_runtime.h>
#include <math.h>

// ---------------------------------------------------------------------------
// PRDC precision loss for MI455X (gfx1250, wave32, WMMA).
// real, fake: [8192, 1024] fp32.  Output: scalar fp32.
// Strategy: bf16 hi/lo split (bf16x3) GEMMs on v_wmma_f32_16x16x32_bf16,
// fused online top-6 (radii) and fused column-max + sigmoid reduction.
// ---------------------------------------------------------------------------

typedef __attribute__((ext_vector_type(16))) __bf16 v16bf;
typedef __attribute__((ext_vector_type(8)))  __bf16 v8bf;
typedef __attribute__((ext_vector_type(8)))  float  v8f;

#define NREAL 8192
#define NFAKE 8192
#define KDIM  1024
#define MTILE 64          // rows of A staged in LDS per workgroup
#define WAVES 4           // 4 waves * 16 rows = 64
#define NBLK  (NREAL/16)  // 512 column blocks of 16

// ---- operand loaders (layouts per CDNA5 ISA 7.12.2) -----------------------
// A (16x32 bf16): lane l supplies row M=l%16.
//   lanes 0-15 : K = kc+0..7   and kc+16..23
//   lanes 16-31: K = kc+8..15  and kc+24..31
__device__ __forceinline__ v16bf load_a(const __bf16* rowp, int kc, int sel) {
    union { v16bf v; v8bf h[2]; } u;
    u.h[0] = *(const v8bf*)(rowp + kc + sel * 8);
    u.h[1] = *(const v8bf*)(rowp + kc + 16 + sel * 8);
    return u.v;
}
// B (32x16 bf16): lane l supplies column N=l%16 (a feature row).
//   lanes 0-15 : K = kc+0..7 and kc+16..23 ; lanes 16-31: K = kc+8..15 and kc+24..31
__device__ __forceinline__ v16bf load_b(const __bf16* rowp, int kc, int sel) {
    union { v16bf v; v8bf h[2]; } u;
    u.h[0] = *(const v8bf*)(rowp + kc + sel * 8);
    u.h[1] = *(const v8bf*)(rowp + kc + 16 + sel * 8);
    return u.v;
}

// ---------------------------------------------------------------------------
// prep: fp32 -> bf16 hi/lo, fp32 row norms, zero the global accumulator.
// grid = NREAL + NFAKE blocks, 256 threads; each block handles one row.
// ---------------------------------------------------------------------------
__global__ __launch_bounds__(256)
void prep_kernel(const float* __restrict__ rf, const float* __restrict__ ff,
                 __bf16* __restrict__ rhi, __bf16* __restrict__ rlo,
                 __bf16* __restrict__ fhi, __bf16* __restrict__ flo,
                 float* __restrict__ rnorm, float* __restrict__ fnorm,
                 float* __restrict__ gsum) {
    __shared__ float red[256];
    const int row = blockIdx.x;
    const float* src;  __bf16 *hi, *lo;  float* nrm;  int r;
    if (row < NREAL) { src = rf; hi = rhi; lo = rlo; nrm = rnorm; r = row; }
    else             { src = ff; hi = fhi; lo = flo; nrm = fnorm; r = row - NREAL; }
    const size_t base = (size_t)r * KDIM;
    float s = 0.0f;
#pragma unroll
    for (int j = 0; j < KDIM / 256; ++j) {
        const int k = threadIdx.x + j * 256;
        const float x  = src[base + k];
        const __bf16 h = (__bf16)x;
        const __bf16 l = (__bf16)(x - (float)h);
        hi[base + k] = h;
        lo[base + k] = l;
        s += x * x;
    }
    red[threadIdx.x] = s;
    __syncthreads();
    for (int off = 128; off > 0; off >>= 1) {
        if (threadIdx.x < off) red[threadIdx.x] += red[threadIdx.x + off];
        __syncthreads();
    }
    if (threadIdx.x == 0) nrm[r] = red[0];
    if (row == 0 && threadIdx.x == 0) *gsum = 0.0f;
}

// ---------------------------------------------------------------------------
// Fused GEMM. MODE 0: A=real, top-6 smallest d^2 per row -> radii.
//            MODE 1: A=fake, max(radii - d) per row -> atomic sigmoid sum.
// Block = 128 threads (4 waves); each wave owns 16 A-rows. A-block (hi+lo)
// staged once in LDS (256 KB of the 320 KB WGP LDS); B streamed from L2.
// ---------------------------------------------------------------------------
template <int MODE>
__global__ __launch_bounds__(128)
void gemm_kernel(const __bf16* __restrict__ Ahi, const __bf16* __restrict__ Alo,
                 const __bf16* __restrict__ Bhi, const __bf16* __restrict__ Blo,
                 const float* __restrict__ normA, const float* __restrict__ normB,
                 float* __restrict__ radii, float* __restrict__ gsum) {
    __shared__ __bf16 sAhi[MTILE * KDIM];   // 128 KB
    __shared__ __bf16 sAlo[MTILE * KDIM];   // 128 KB
    __shared__ float  sScr[WAVES][16 * 16]; // 4 KB (mode 0 transpose scratch)

    const int tid  = threadIdx.x;
    const int m0   = blockIdx.x * MTILE;

    // cooperative stage of A block (16B vectors)
    {
        const uint4* gh = (const uint4*)(Ahi + (size_t)m0 * KDIM);
        const uint4* gl = (const uint4*)(Alo + (size_t)m0 * KDIM);
        uint4* sh = (uint4*)sAhi;
        uint4* sl = (uint4*)sAlo;
        for (int i = tid; i < MTILE * KDIM / 8; i += 128) { sh[i] = gh[i]; sl[i] = gl[i]; }
    }
    __syncthreads();

    const int wave = tid >> 5;
    const int lane = tid & 31;
    const int sel  = lane >> 4;   // half-wave select
    const int lcol = lane & 15;   // C-tile column / A row index
    const int mw   = wave * 16;   // local row base of this wave

    float rn[8];                  // norms of this lane's 8 accumulator rows
#pragma unroll
    for (int v = 0; v < 8; ++v) rn[v] = normA[m0 + mw + v + 8 * sel];

    float t6[6];                  // mode0: 6 smallest d^2, ascending
#pragma unroll
    for (int i = 0; i < 6; ++i) t6[i] = 1e30f;
    float rowmax[8];              // mode1: running max(radius - d)
#pragma unroll
    for (int v = 0; v < 8; ++v) rowmax[v] = -1e30f;

    const __bf16* arow_h = sAhi + (size_t)(mw + lcol) * KDIM;
    const __bf16* arow_l = sAlo + (size_t)(mw + lcol) * KDIM;

    for (int nb = 0; nb < NBLK; ++nb) {
        const int ncol  = nb * 16 + lcol;
        const float cn  = normB[ncol];
        float crad = 0.0f;
        if (MODE == 1) crad = radii[ncol];
        const __bf16* brow_h = Bhi + (size_t)ncol * KDIM;
        const __bf16* brow_l = Blo + (size_t)ncol * KDIM;

        v8f acc0 = {0.f, 0.f, 0.f, 0.f, 0.f, 0.f, 0.f, 0.f};
        v8f acc1 = {0.f, 0.f, 0.f, 0.f, 0.f, 0.f, 0.f, 0.f};
#pragma unroll 4
        for (int kc = 0; kc < KDIM; kc += 32) {
            const v16bf aH = load_a(arow_h, kc, sel);
            const v16bf aL = load_a(arow_l, kc, sel);
            const v16bf bH = load_b(brow_h, kc, sel);
            const v16bf bL = load_b(brow_l, kc, sel);
            // bf16x3: hi*hi into acc0; hi*lo + lo*hi into acc1 (separate chains)
            acc0 = __builtin_amdgcn_wmma_f32_16x16x32_bf16(false, aH, false, bH,
                                                           (short)0, acc0, false, false);
            acc1 = __builtin_amdgcn_wmma_f32_16x16x32_bf16(false, aH, false, bL,
                                                           (short)0, acc1, false, false);
            acc1 = __builtin_amdgcn_wmma_f32_16x16x32_bf16(false, aL, false, bH,
                                                           (short)0, acc1, false, false);
        }

        if (MODE == 0) {
            // d^2 tile -> per-wave LDS scratch to hand each row to a lane pair (l, l^16)
            float* scr = sScr[wave];
#pragma unroll
            for (int v = 0; v < 8; ++v) {
                const float dot = acc0[v] + acc1[v];
                float d2 = rn[v] + cn - 2.0f * dot;
                d2 = fmaxf(d2, 0.0f) + 1e-8f;
                scr[(v + 8 * sel) * 16 + lcol] = d2;
            }
            __builtin_amdgcn_wave_barrier();
            // lane l scans row (l&15), columns sel*8..sel*8+7 of the tile
#pragma unroll
            for (int c = 0; c < 8; ++c) {
                float val = scr[lcol * 16 + sel * 8 + c];
#pragma unroll
                for (int i = 0; i < 6; ++i) {
                    const float lo = fminf(t6[i], val);
                    const float hi = fmaxf(t6[i], val);
                    t6[i] = lo; val = hi;
                }
            }
            __builtin_amdgcn_wave_barrier();
        } else {
            // row = fake index, col = real index: val = radii_col - d(row,col)
#pragma unroll
            for (int v = 0; v < 8; ++v) {
                const float dot = acc0[v] + acc1[v];
                float d2 = rn[v] + cn - 2.0f * dot;
                const float d = sqrtf(fmaxf(d2, 0.0f) + 1e-8f);
                rowmax[v] = fmaxf(rowmax[v], crad - d);
            }
        }
    }

    if (MODE == 0) {
        // merge the two half-wave top-6 lists for row lcol (partner = lane^16)
#pragma unroll
        for (int i = 0; i < 6; ++i) {
            float val = __shfl_xor(t6[i], 16, 32);
#pragma unroll
            for (int j = 0; j < 6; ++j) {
                const float lo = fminf(t6[j], val);
                const float hi = fmaxf(t6[j], val);
                t6[j] = lo; val = hi;
            }
        }
        if (sel == 0) radii[m0 + mw + lcol] = sqrtf(t6[5]); // 6th smallest (k=5)
    } else {
        float lsum = 0.0f;
#pragma unroll
        for (int v = 0; v < 8; ++v) {
            float m = rowmax[v];
            m = fmaxf(m, __shfl_xor(m, 1, 32));
            m = fmaxf(m, __shfl_xor(m, 2, 32));
            m = fmaxf(m, __shfl_xor(m, 4, 32));
            m = fmaxf(m, __shfl_xor(m, 8, 32));
            // sigmoid(m / TEMP), TEMP = 0.1
            lsum += 1.0f / (1.0f + expf(-m * 10.0f));
        }
        if (lcol == 0) atomicAdd(gsum, lsum); // lanes 0 and 16: 8 rows each
    }
}

__global__ void finalize_kernel(const float* __restrict__ gsum, float* __restrict__ out) {
    out[0] = -(gsum[0] * (1.0f / (float)NFAKE));
}

// ---------------------------------------------------------------------------
extern "C" void kernel_launch(void* const* d_in, const int* in_sizes, int n_in,
                              void* d_out, int out_size, void* d_ws, size_t ws_size,
                              hipStream_t stream) {
    const float* real_f = (const float*)d_in[0];
    const float* fake_f = (const float*)d_in[1];
    float* out = (float*)d_out;

    char* ws = (char*)d_ws;
    const size_t feat = (size_t)NREAL * KDIM;
    __bf16* rhi = (__bf16*)ws;                 ws += feat * sizeof(__bf16);
    __bf16* rlo = (__bf16*)ws;                 ws += feat * sizeof(__bf16);
    __bf16* fhi = (__bf16*)ws;                 ws += feat * sizeof(__bf16);
    __bf16* flo = (__bf16*)ws;                 ws += feat * sizeof(__bf16);
    float* rnorm = (float*)ws;                 ws += NREAL * sizeof(float);
    float* fnorm = (float*)ws;                 ws += NFAKE * sizeof(float);
    float* radii = (float*)ws;                 ws += NREAL * sizeof(float);
    float* gsum  = (float*)ws;                 ws += sizeof(float);

    prep_kernel<<<NREAL + NFAKE, 256, 0, stream>>>(real_f, fake_f, rhi, rlo, fhi, flo,
                                                   rnorm, fnorm, gsum);
    // Stage 1: radii from real x real^T (top-6 smallest per row)
    gemm_kernel<0><<<NREAL / MTILE, 32 * WAVES, 0, stream>>>(
        rhi, rlo, rhi, rlo, rnorm, rnorm, radii, gsum);
    // Stage 2: fake x real^T, fused max + sigmoid reduction
    gemm_kernel<1><<<NFAKE / MTILE, 32 * WAVES, 0, stream>>>(
        fhi, flo, rhi, rlo, fnorm, rnorm, radii, gsum);
    finalize_kernel<<<1, 1, 0, stream>>>(gsum, out);
}
